// EvaluationModel_28570122453776
// MI455X (gfx1250) — matrix operand order repaired and checked by reading, not compiled, not run on hardware
//
#include <hip/hip_runtime.h>

// ---------------------------------------------------------------------------
// D-ETM temporal attention scan for MI455X (gfx1250), bf16 WMMA pipeline.
//   K=128 topics, T=256 steps, E=320, H=10 heads, HD=32 (= WMMA K-dim).
// All tensors (~45MB) are L2-resident (192MB L2), so fragments load straight
// from global; the scan is latency-bound, so kernels are short and WMMA-dense.
// Attention uses split-KV flash (4 splits) to raise occupancy 80->320 waves.
// ---------------------------------------------------------------------------

#define KTOP   128
#define TT     256
#define EE     320
#define HH     10
#define HD     32
#define TM1    255
#define SMAX   (TM1 * KTOP)      // 32640 max KV rows
#define NSPLIT 4                 // KV splits; S/4 is always a multiple of 32
#define LNEPS  1e-5f
#define QSCALE 0.17677669529663687f   // 1/sqrt(HD)

typedef unsigned short u16;
typedef unsigned int   u32;

typedef __bf16 v16bf __attribute__((ext_vector_type(16)));
typedef float  v8f   __attribute__((ext_vector_type(8)));

union FragU { v16bf v; uint4 q[2]; };

__device__ __forceinline__ u16 f2bf(float x) {
  u32 u = __float_as_uint(x);
  u32 r = u + 0x7FFFu + ((u >> 16) & 1u);   // round-to-nearest-even
  return (u16)(r >> 16);
}

// A-matrix fragment (16x32 bf16): lane L holds row m = m0 + (L&15);
// elements 0..7  = A[m, k0 + 8*(L>>4) .. +7]
// elements 8..15 = A[m, k0 + 16 + 8*(L>>4) .. +7]   (two contiguous 16B chunks)
__device__ __forceinline__ v16bf load_frag_a(const u16* base, int m0, int k0,
                                             long ld, int lane) {
  int m = m0 + (lane & 15);
  int half = lane >> 4;
  const u16* p = base + (long)m * ld + k0 + half * 8;
  FragU f;
  f.q[0] = *(const uint4*)(p);
  f.q[1] = *(const uint4*)(p + 16);
  return f.v;
}

// B-matrix fragment (32x16 bf16), B[k,n] stored as rows of `base` (= W[n,k]):
// lane L holds column n = n0 + (L&15); elements 0..7 = base[n, k0 + 8*(L>>4) .. +7],
// elements 8..15 = base[n, k0 + 16 + 8*(L>>4) .. +7]  (two 16B chunks, matches A).
__device__ __forceinline__ v16bf load_frag_b(const u16* base, int n0, int k0,
                                             long ld, int lane) {
  int n = n0 + (lane & 15);
  int half = lane >> 4;
  const u16* p = base + (long)n * ld + k0 + half * 8;
  FragU f;
  f.q[0] = *(const uint4*)(p);
  f.q[1] = *(const uint4*)(p + 16);
  return f.v;
}

__device__ __forceinline__ v8f wmma_bf16(v16bf a, v16bf b, v8f c) {
  return __builtin_amdgcn_wmma_f32_16x16x32_bf16(false, a, false, b,
                                                 (short)0, c, false, false);
}

// ---------------------------------------------------------------------------
// One-time setup: fold MHA in-proj into the qkv_proj weights.
//   q2 = alpha @ (Wq@Qw)^T + (Wq@bq1 + bq)   (scale folded into q weights)
// Also converts out_w to bf16.
// idx layout: [0,3EE) wcomb | [3EE,4EE) outw | [4EE,4EE+3E) bcomb
// ---------------------------------------------------------------------------
__global__ void k_setup(const float* __restrict__ qkv_w,
                        const float* __restrict__ qkv_b,
                        const float* __restrict__ in_w,
                        const float* __restrict__ in_b,
                        const float* __restrict__ out_w,
                        u16* __restrict__ wcomb, float* __restrict__ bcomb,
                        u16* __restrict__ outw_bf) {
  const int ESQ = EE * EE;
  int idx = blockIdx.x * blockDim.x + threadIdx.x;
  if (idx < 3 * ESQ) {
    int mat = idx / ESQ, rem = idx % ESQ;
    int i = rem / EE, j = rem % EE;
    const float* wa = in_w + (size_t)(mat * EE + i) * EE;  // row i of W_{q/k/v}
    const float* wb = qkv_w + (size_t)(mat * EE) * EE + j; // col j of qkv block
    float acc = 0.f;
    for (int k = 0; k < EE; ++k) acc += wa[k] * wb[(size_t)k * EE];
    if (mat == 0) acc *= QSCALE;
    wcomb[idx] = f2bf(acc);
  } else if (idx < 4 * ESQ) {
    int rem = idx - 3 * ESQ;
    outw_bf[rem] = f2bf(out_w[rem]);
  } else if (idx < 4 * ESQ + 3 * EE) {
    int r = idx - 4 * ESQ;            // r in [0, 3E)
    int mat = r / EE;
    const float* wa = in_w + (size_t)r * EE;
    float acc = in_b[r];
    for (int k = 0; k < EE; ++k) acc += wa[k] * qkv_b[mat * EE + k];
    if (mat == 0) acc *= QSCALE;
    bcomb[r] = acc;
  }
}

// ---------------------------------------------------------------------------
// alpha0 = LayerNorm(mu_0_alpha);  writes d_out[t=0] and working alpha bufs.
// grid = K blocks of E threads.
// ---------------------------------------------------------------------------
__global__ void k_ln0(const float* __restrict__ mu0,
                      const float* __restrict__ g, const float* __restrict__ b,
                      float* __restrict__ out, float* __restrict__ alpha_f,
                      u16* __restrict__ alpha_bf) {
  __shared__ float ssum[EE / 32], ssq[EE / 32], smv[2];
  int row = blockIdx.x, t = threadIdx.x;
  float x = mu0[(size_t)row * EE + t];
  float s = x, s2 = x * x;
  for (int off = 16; off; off >>= 1) {
    s  += __shfl_down(s, off, 32);
    s2 += __shfl_down(s2, off, 32);
  }
  if ((t & 31) == 0) { ssum[t >> 5] = s; ssq[t >> 5] = s2; }
  __syncthreads();
  if (t == 0) {
    float a = 0.f, c = 0.f;
    for (int i = 0; i < EE / 32; ++i) { a += ssum[i]; c += ssq[i]; }
    float m = a / EE;
    smv[0] = m;
    smv[1] = rsqrtf(c / EE - m * m + LNEPS);
  }
  __syncthreads();
  float y = (x - smv[0]) * smv[1] * g[t] + b[t];
  out[(size_t)row * EE + t]     = y;
  alpha_f[(size_t)row * EE + t] = y;
  alpha_bf[(size_t)row * EE + t] = f2bf(y);
}

// ---------------------------------------------------------------------------
// Per-step projections: q2 / k2 / v2 = alpha @ Wcomb^T + bcomb (scale folded).
// k2 -> kcache row block t (S x E); v2 -> transposed vT cache (E x SMAX) so
// the P*V GEMM's B-fragments are contiguous. 480 wave-tiles = 60 blocks x 8.
// ---------------------------------------------------------------------------
__global__ __launch_bounds__(256) void k_proj(
    const u16* __restrict__ alpha_bf, const u16* __restrict__ wcomb,
    const float* __restrict__ bcomb, u16* __restrict__ q2,
    u16* __restrict__ kcache, u16* __restrict__ vT, int t) {
  int lane = threadIdx.x & 31;
  int wave = (blockIdx.x * blockDim.x + threadIdx.x) >> 5;  // 0..479
  int mat = wave / 160, rem = wave % 160;
  int m0 = (rem / 20) * 16, n0 = (rem % 20) * 16;
  const u16* wb = wcomb + (size_t)mat * EE * EE;

  v8f c = {};
  for (int kk = 0; kk < EE; kk += 32) {
    v16bf a = load_frag_a(alpha_bf, m0, kk, EE, lane);
    v16bf b = load_frag_b(wb, n0, kk, EE, lane);
    c = wmma_bf16(a, b, c);
  }
  int n = n0 + (lane & 15), half = lane >> 4;
  float bias = bcomb[mat * EE + n];
  for (int r = 0; r < 8; ++r) {
    int m = m0 + r + 8 * half;          // C layout: row = r + 8*half
    u16 h = f2bf(c[r] + bias);
    if (mat == 0)       q2[(size_t)m * EE + n] = h;
    else if (mat == 1)  kcache[(size_t)(t * KTOP + m) * EE + n] = h;
    else                vT[(size_t)n * SMAX + t * KTOP + m] = h;
  }
}

// ---------------------------------------------------------------------------
// Split-KV flash attention: one wave per (head, 16-row q-tile, kv-split);
// 320 waves = 40 blocks x 8. Each split covers S/4 keys (always a multiple
// of 32). HD=32 => one WMMA per 16x16 score tile. Online softmax via 16-lane
// shfl_xor reductions (C-tile: row = r+8*half, col = lane&15). P-tile is
// bounced through a private 1KB LDS region (C-layout -> A-layout).
// Emits unnormalized o partials + per-row (max, sumexp) for the combiner.
// ---------------------------------------------------------------------------
__global__ __launch_bounds__(256) void k_attn(
    const u16* __restrict__ q2, const u16* __restrict__ kcache,
    const u16* __restrict__ vT, float* __restrict__ o_part,
    float* __restrict__ ml_part, int t) {
  __shared__ __align__(16) u16 pbuf[8][16 * 32];
  int lane = threadIdx.x & 31, wid = threadIdx.x >> 5;
  int task = blockIdx.x * 8 + wid;          // 0..319
  int head = task >> 5;                     // 10 heads
  int rem  = task & 31;
  int mtile = rem >> 2, split = rem & 3;    // 8 q-tiles x 4 kv-splits
  int m0 = mtile * 16, h32 = head * HD;
  int S = (t + 1) * KTOP;
  int slen = S / NSPLIT;                    // multiple of 32
  int sbeg = split * slen, send = sbeg + slen;
  int half = lane >> 4, col = lane & 15;
  u16* pb = pbuf[wid];

  v16bf aq = load_frag_a(q2, m0, h32, EE, lane);
  v8f acc0 = {}, acc1 = {};
  float mrow[8], lrow[8];
  for (int r = 0; r < 8; ++r) { mrow[r] = -__builtin_inff(); lrow[r] = 0.f; }

  for (int s0 = sbeg; s0 < send; s0 += 32) {
    v8f z = {};
    v16bf bk0 = load_frag_b(kcache, s0,      h32, EE, lane);
    v8f c0 = wmma_bf16(aq, bk0, z);
    v16bf bk1 = load_frag_b(kcache, s0 + 16, h32, EE, lane);
    v8f c1 = wmma_bf16(aq, bk1, z);

    float p0[8], p1[8];
    for (int r = 0; r < 8; ++r) {
      float cm = fmaxf(c0[r], c1[r]);
      for (int off = 1; off < 16; off <<= 1)
        cm = fmaxf(cm, __shfl_xor(cm, off, 16));
      float mnew = fmaxf(mrow[r], cm);
      float sc = __expf(mrow[r] - mnew);       // 0 on first chunk (-inf)
      p0[r] = __expf(c0[r] - mnew);
      p1[r] = __expf(c1[r] - mnew);
      float rs = p0[r] + p1[r];
      for (int off = 1; off < 16; off <<= 1)
        rs += __shfl_xor(rs, off, 16);
      lrow[r] = lrow[r] * sc + rs;
      acc0[r] *= sc;  acc1[r] *= sc;
      mrow[r] = mnew;
    }

    // C-layout -> row-major P tile [16][32] in wave-private LDS.
    asm volatile("" ::: "memory");
    for (int r = 0; r < 8; ++r) {
      int row = r + 8 * half;
      pb[row * 32 + col]      = f2bf(p0[r]);
      pb[row * 32 + 16 + col] = f2bf(p1[r]);
    }
    asm volatile("" ::: "memory");          // DS pipe is in-order per wave
    v16bf ap = load_frag_a(pb, 0, 0, 32, lane);
    asm volatile("" ::: "memory");

    v16bf bv0 = load_frag_b(vT, h32,      s0, (long)SMAX, lane);
    v16bf bv1 = load_frag_b(vT, h32 + 16, s0, (long)SMAX, lane);
    acc0 = wmma_bf16(ap, bv0, acc0);
    acc1 = wmma_bf16(ap, bv1, acc1);
  }

  for (int r = 0; r < 8; ++r) {
    int m = m0 + r + 8 * half;
    o_part[((size_t)split * KTOP + m) * EE + h32 + col]      = acc0[r];
    o_part[((size_t)split * KTOP + m) * EE + h32 + 16 + col] = acc1[r];
    if (col == 0) {                         // row stats (replicated per group)
      size_t si = ((size_t)(split * HH + head) * KTOP + m) * 2;
      ml_part[si]     = mrow[r];
      ml_part[si + 1] = lrow[r];
    }
  }
}

// ---------------------------------------------------------------------------
// Combine the 4 KV-split partials with exact log-sum-exp rescale:
//   M = max_i m_i;  L = sum_i e^{m_i-M} l_i;  o = (sum_i e^{m_i-M} o_i) / L
// One thread per (m, e) element: 160 blocks x 256.
// ---------------------------------------------------------------------------
__global__ __launch_bounds__(256) void k_attn_comb(
    const float* __restrict__ o_part, const float* __restrict__ ml_part,
    u16* __restrict__ o_bf) {
  int idx = blockIdx.x * 256 + threadIdx.x;   // 0 .. K*E-1
  if (idx >= KTOP * EE) return;
  int m = idx / EE, e = idx % EE;
  int head = e / HD;
  float ms[NSPLIT], ls[NSPLIT];
  float M = -__builtin_inff();
  for (int i = 0; i < NSPLIT; ++i) {
    size_t si = ((size_t)(i * HH + head) * KTOP + m) * 2;
    ms[i] = ml_part[si];
    ls[i] = ml_part[si + 1];
    M = fmaxf(M, ms[i]);
  }
  float L = 0.f, val = 0.f;
  for (int i = 0; i < NSPLIT; ++i) {
    float w = __expf(ms[i] - M);
    L   += w * ls[i];
    val += w * o_part[((size_t)i * KTOP + m) * EE + e];
  }
  o_bf[idx] = f2bf(val / L);
}

// ---------------------------------------------------------------------------
// Out-proj + residual + LayerNorm -> new alpha (and d_out row t+1).
// 8 blocks x 320 threads (10 waves); each block owns a 16-row tile so the
// full E=320 row is available in LDS for the LN reduction.
// ---------------------------------------------------------------------------
__global__ __launch_bounds__(320) void k_outln(
    const u16* __restrict__ o_bf, const u16* __restrict__ outw_bf,
    const float* __restrict__ out_b, const float* __restrict__ ln_g,
    const float* __restrict__ ln_b, float* __restrict__ alpha_f,
    u16* __restrict__ alpha_bf, float* __restrict__ d_out, int t) {
  __shared__ float ybuf[16 * EE];
  __shared__ float stats[16][2];
  int tid = threadIdx.x, wid = tid >> 5, lane = tid & 31;
  int m0 = blockIdx.x * 16;
  int half = lane >> 4, colL = lane & 15;

  for (int nt = wid; nt < 20; nt += 10) {     // 2 n-tiles per wave
    int n0 = nt * 16;
    v8f c = {};
    for (int kk = 0; kk < EE; kk += 32) {
      v16bf a = load_frag_a(o_bf, m0, kk, EE, lane);
      v16bf b = load_frag_b(outw_bf, n0, kk, EE, lane);
      c = wmma_bf16(a, b, c);
    }
    for (int r = 0; r < 8; ++r)
      ybuf[(r + 8 * half) * EE + n0 + colL] = c[r];
  }
  __syncthreads();

  if (tid < 16) {
    const float* yr = ybuf + tid * EE;
    const float* ar = alpha_f + (size_t)(m0 + tid) * EE;
    float s = 0.f, s2 = 0.f;
    for (int j = 0; j < EE; ++j) {
      float v = yr[j] + out_b[j] + ar[j];
      s += v; s2 += v * v;
    }
    float m = s / EE;
    stats[tid][0] = m;
    stats[tid][1] = rsqrtf(s2 / EE - m * m + LNEPS);
  }
  __syncthreads();

  int j = tid;                                 // column j, all 320 threads
  for (int row = 0; row < 16; ++row) {
    int gm = m0 + row;
    float v = ybuf[row * EE + j] + out_b[j] + alpha_f[(size_t)gm * EE + j];
    float y = (v - stats[row][0]) * stats[row][1] * ln_g[j] + ln_b[j];
    d_out[((size_t)(t + 1) * KTOP + gm) * EE + j] = y;
    alpha_f[(size_t)gm * EE + j]  = y;
    alpha_bf[(size_t)gm * EE + j] = f2bf(y);
  }
}

// ---------------------------------------------------------------------------
extern "C" void kernel_launch(void* const* d_in, const int* in_sizes, int n_in,
                              void* d_out, int out_size, void* d_ws,
                              size_t ws_size, hipStream_t stream) {
  (void)in_sizes; (void)n_in; (void)out_size; (void)ws_size;
  const float* mu0    = (const float*)d_in[0];
  const float* qkv_w  = (const float*)d_in[1];
  const float* qkv_b  = (const float*)d_in[2];
  const float* in_w   = (const float*)d_in[3];
  const float* in_b   = (const float*)d_in[4];
  const float* out_w  = (const float*)d_in[5];
  const float* out_b  = (const float*)d_in[6];
  const float* ln_g   = (const float*)d_in[7];
  const float* ln_b   = (const float*)d_in[8];
  float* out = (float*)d_out;

  char* ws = (char*)d_ws;
  size_t off = 0;
  auto alloc = [&](size_t bytes) -> void* {
    void* p = ws + off;
    off += (bytes + 255) & ~(size_t)255;
    return p;
  };
  u16*   wcomb    = (u16*)alloc((size_t)3 * EE * EE * 2);   // folded qkv weights
  u16*   outw_bf  = (u16*)alloc((size_t)EE * EE * 2);
  float* bcomb    = (float*)alloc((size_t)3 * EE * 4);
  float* alpha_f  = (float*)alloc((size_t)KTOP * EE * 4);
  u16*   alpha_bf = (u16*)alloc((size_t)KTOP * EE * 2);
  u16*   q2       = (u16*)alloc((size_t)KTOP * EE * 2);
  u16*   o_bf     = (u16*)alloc((size_t)KTOP * EE * 2);
  float* o_part   = (float*)alloc((size_t)NSPLIT * KTOP * EE * 4);
  float* ml_part  = (float*)alloc((size_t)NSPLIT * HH * KTOP * 2 * 4);
  u16*   kcache   = (u16*)alloc((size_t)SMAX * EE * 2);     // (S x E)
  u16*   vT       = (u16*)alloc((size_t)EE * SMAX * 2);     // (E x S) transposed
  // total ~44MB scratch; whole working set is L2-resident on MI455X.

  int setup_threads = 4 * EE * EE + 3 * EE;
  k_setup<<<(setup_threads + 255) / 256, 256, 0, stream>>>(
      qkv_w, qkv_b, in_w, in_b, out_w, wcomb, bcomb, outw_bf);
  k_ln0<<<KTOP, EE, 0, stream>>>(mu0, ln_g, ln_b, out, alpha_f, alpha_bf);

  for (int t = 0; t < TM1; ++t) {
    k_proj<<<60, 256, 0, stream>>>(alpha_bf, wcomb, bcomb, q2, kcache, vT, t);
    k_attn<<<40, 256, 0, stream>>>(q2, kcache, vT, o_part, ml_part, t);
    k_attn_comb<<<(KTOP * EE + 255) / 256, 256, 0, stream>>>(o_part, ml_part,
                                                             o_bf);
    k_outln<<<8, 320, 0, stream>>>(o_bf, outw_bf, out_b, ln_g, ln_b,
                                   alpha_f, alpha_bf, out, t);
  }
}
